// TreeConv1d_67808943669269
// MI455X (gfx1250) — compile-verified, matrix operand order checked by reading, not run on hardware
//
#include <hip/hip_runtime.h>

#define B_ 128
#define CIN 256
#define NPTS 1024
#define COUT 256
#define LIDX 3069
#define NCOL 1023
#define KCHUNKS 24   // 768 / 32
#define XROW 34      // padded row pitch (bf16 elems) to avoid LDS bank conflicts

typedef __attribute__((ext_vector_type(16))) __bf16        v16bf;
typedef __attribute__((ext_vector_type(8)))  float         v8f;
typedef __attribute__((ext_vector_type(8)))  unsigned int  u32x8;

__device__ __forceinline__ unsigned short f2bf(float f) {
  unsigned u = __builtin_bit_cast(unsigned, f);
  u += 0x7FFFu + ((u >> 16) & 1u);          // round-to-nearest-even
  return (unsigned short)(u >> 16);
}

// Build bf16 A-fragments from fp32 weight, pre-swizzled to the CDNA5
// 16-bit A(16x32) layout. K-flattening: kg = k*256 + c.
__global__ void prep_wfrag(const float* __restrict__ weight,
                           unsigned short* __restrict__ wfrag) {
  int tid  = blockIdx.x * 256 + threadIdx.x;   // 196608 total = 256*768
  int j    = tid & 15;
  int lane = (tid >> 4) & 31;
  int rest = tid >> 9;                          // mt*24 + kc
  int kc   = rest % 24;
  int mt   = rest / 24;
  int m    = mt * 16 + (lane & 15);
  int kb   = (lane >> 4) << 3;                  // 0 or 8
  int kl   = (j < 8) ? (kb + j) : (16 + kb + (j - 8));
  int kg   = kc * 32 + kl;
  int k    = kg >> 8;
  int c    = kg & 255;
  wfrag[tid] = f2bf(weight[(m * 256 + c) * 3 + k]);
}

__global__ void zero_col0(float* __restrict__ out) {
  size_t o = (size_t)blockIdx.x * 256 + threadIdx.x;  // b*256 + o
  out[o * (size_t)NPTS] = 0.0f;
}

__global__ __launch_bounds__(256)
void tree_gemm(const float* __restrict__ data, const long long* __restrict__ idx,
               const unsigned short* __restrict__ wfrag,
               const float* __restrict__ bias, float* __restrict__ out) {
  __shared__ int sIdx[192];
  __shared__ __align__(16) unsigned short sX[2][64 * XROW];  // double-buffered X tile

  const int b  = blockIdx.x >> 4;
  const int nt = blockIdx.x & 15;
  const int n0 = nt * 64;
  const int t  = threadIdx.x;

  if (t < 192) {
    int gi = n0 * 3 + t;
    sIdx[t] = (gi < LIDX) ? (int)idx[(size_t)b * LIDX + gi] : 0;
  }
  __syncthreads();

  const int lane  = t & 31;
  const int w     = t >> 5;              // wave 0..7 -> rows [w*32, w*32+32)
  const int laneM = lane & 15;
  const int kb    = (lane >> 4) << 3;    // 0 or 8 (K-half select / M+8 select)

  const float* dataB = data + (size_t)b * CIN * NPTS;

  v8f acc[2][4];
  #pragma unroll
  for (int h = 0; h < 2; ++h)
    #pragma unroll
    for (int nc = 0; nc < 4; ++nc)
      #pragma unroll
      for (int g = 0; g < 8; ++g) acc[h][nc][g] = 0.0f;

  float bv[2][8];
  #pragma unroll
  for (int h = 0; h < 2; ++h)
    #pragma unroll
    for (int g = 0; g < 8; ++g)
      bv[h][g] = bias[w * 32 + h * 16 + g + kb];

  const int gn   = t >> 2;          // gather column 0..63
  const int gkl0 = (t & 3) << 3;    // gather K-base 0,8,16,24

  // Stage the gathered 32x64 bf16 X tile for K-chunk kc into `buf`.
  auto stage = [&](int kc, unsigned short* buf) {
    const int kk = kc >> 3;                 // conv tap 0..2
    const int c0 = (kc & 7) << 5;           // channel base
    const int iv = sIdx[gn * 3 + kk];
    const float* dp = dataB + (size_t)(c0 + gkl0) * NPTS + iv;
    unsigned int* xp = (unsigned int*)&buf[gn * XROW + gkl0];
    #pragma unroll
    for (int i = 0; i < 4; ++i) {
      float v0 = dp[(size_t)(2 * i)     * NPTS];
      float v1 = dp[(size_t)(2 * i + 1) * NPTS];
      xp[i] = (unsigned)f2bf(v0) | ((unsigned)f2bf(v1) << 16);
    }
  };

  stage(0, sX[0]);
  __syncthreads();

  for (int kc = 0; kc < KCHUNKS; ++kc) {
    const unsigned short* cur = sX[kc & 1];

    // Prefetch-gather the next K-chunk into the alternate buffer; these
    // global loads + ds_stores overlap the WMMAs below.
    if (kc + 1 < KCHUNKS) stage(kc + 1, sX[(kc + 1) & 1]);

    // A fragments: pre-swizzled, one 32B vector load each.
    const u32x8 a0 = *(const u32x8*)(wfrag + (size_t)(((2 * w)     * 24 + kc) * 32 + lane) * 16);
    const u32x8 a1 = *(const u32x8*)(wfrag + (size_t)(((2 * w + 1) * 24 + kc) * 32 + lane) * 16);
    const v16bf A0 = __builtin_bit_cast(v16bf, a0);
    const v16bf A1 = __builtin_bit_cast(v16bf, a1);

    #pragma unroll
    for (int nc = 0; nc < 4; ++nc) {
      const unsigned short* rowp = &cur[(nc * 16 + laneM) * XROW + kb];
      u32x8 br;
      #pragma unroll
      for (int i = 0; i < 4; ++i) {
        br[i]     = *(const unsigned int*)(rowp + 2 * i);        // K = kb+2i, kb+2i+1
        br[i + 4] = *(const unsigned int*)(rowp + 16 + 2 * i);   // K = 16+kb+2i, +1
      }
      const v16bf Bf = __builtin_bit_cast(v16bf, br);
      acc[0][nc] = __builtin_amdgcn_wmma_f32_16x16x32_bf16(
          false, A0, false, Bf, (short)0, acc[0][nc], false, false);
      acc[1][nc] = __builtin_amdgcn_wmma_f32_16x16x32_bf16(
          false, A1, false, Bf, (short)0, acc[1][nc], false, false);
    }

    // Single barrier per iteration: separates this chunk's B-reads from the
    // next-next chunk's writes, and next chunk's writes from its reads.
    if (kc + 1 < KCHUNKS) __syncthreads();
  }

  // ---- store: C layout = VGPR g -> row g (+8 for lanes>=16), lane&15 -> col ----
  #pragma unroll
  for (int h = 0; h < 2; ++h) {
    const int row0 = w * 32 + h * 16 + kb;
    #pragma unroll
    for (int nc = 0; nc < 4; ++nc) {
      const int ng = n0 + nc * 16 + laneM;
      if (ng < NCOL) {
        float* op = out + ((size_t)b * COUT + row0) * NPTS + 1 + ng;
        #pragma unroll
        for (int g = 0; g < 8; ++g)
          op[(size_t)g * NPTS] = acc[h][nc][g] + bv[h][g];
      }
    }
  }
}

extern "C" void kernel_launch(void* const* d_in, const int* in_sizes, int n_in,
                              void* d_out, int out_size, void* d_ws, size_t ws_size,
                              hipStream_t stream) {
  const float*     data   = (const float*)d_in[0];
  const long long* idx    = (const long long*)d_in[1];
  const float*     weight = (const float*)d_in[2];
  const float*     bias   = (const float*)d_in[3];
  float* out = (float*)d_out;
  unsigned short* wfrag = (unsigned short*)d_ws;   // 393216 bytes

  prep_wfrag<<<768, 256, 0, stream>>>(weight, wfrag);
  zero_col0<<<B_, 256, 0, stream>>>(out);
  tree_gemm<<<B_ * 16, 256, 0, stream>>>(data, idx, wfrag, bias, out);

  // Second tuple output: indexes passthrough (int64), raw bytes after feats.
  const size_t featBytes = (size_t)B_ * COUT * NPTS * sizeof(float);
  hipMemcpyAsync((char*)d_out + featBytes, d_in[1],
                 (size_t)B_ * LIDX * sizeof(long long),
                 hipMemcpyDeviceToDevice, stream);
}
